// SimpleRetention_15702400434236
// MI455X (gfx1250) — compile-verified, matrix operand order checked by reading, not run on hardware
//
#include <hip/hip_runtime.h>
#include <hip/hip_bf16.h>

// SimpleRetention for MI455X (gfx1250), compile-only target.
// Pipeline: [WMMA GEMM: Q/K/V proj, bf16 out] -> [xpos] ->
//           [WMMA GEMM batched: S = Q K^T, bf16] -> [softmax * gamma^|q-k|, in place] ->
//           [WMMA GEMM batched: out = P V, f32 out]
// Workspace: Q bf16 32MB | K bf16 32MB | V bf16 32MB | S/P bf16 64MB (160MB total).
// GEMM: 128x128x32 block tile, 8 waves (4x2), double-buffered LDS, bf16 tiles
// staged with GLOBAL_LOAD_ASYNC_TO_LDS_B128 (ASYNCcnt) on gfx1250.

typedef __attribute__((ext_vector_type(16))) __bf16 v16bf;
typedef __attribute__((ext_vector_type(8)))  float  v8f;

// Exact vector/pointer types for the async-LDS builtin (from clang diagnostic:
// param0 is "vector_size(16) int" in the __device__ (addrspace(1)) AS).
typedef int v4i_vs __attribute__((vector_size(16)));
typedef __attribute__((address_space(1))) v4i_vs* gas_v4i_ptr;
typedef __attribute__((address_space(3))) v4i_vs* las_v4i_ptr;

#define BMT 128
#define BNT 128
#define BKT 32
#define LPAD 8
#define LSTR (BKT + LPAD)   // 40 shorts = 80 bytes per LDS row (16B multiple)

#if defined(__has_builtin)
#  if __has_builtin(__builtin_amdgcn_global_load_async_to_lds_b128)
#    define HAVE_ASYNC_LDS 1
#  else
#    define HAVE_ASYNC_LDS 0
#  endif
#  if __has_builtin(__builtin_amdgcn_s_wait_asynccnt)
#    define HAVE_WAIT_ASYNC_BUILTIN 1
#  else
#    define HAVE_WAIT_ASYNC_BUILTIN 0
#  endif
#else
#  define HAVE_ASYNC_LDS 0
#  define HAVE_WAIT_ASYNC_BUILTIN 0
#endif

__device__ __forceinline__ void wait_async_lds() {
#if HAVE_ASYNC_LDS
#  if HAVE_WAIT_ASYNC_BUILTIN
  __builtin_amdgcn_s_wait_asynccnt(0);
#  else
  asm volatile("s_wait_asynccnt 0x0" ::: "memory");
#  endif
#endif
}

// Native bf16 conversion: backend picks v_cvt_pk_bf16_f32 / native cvt on gfx1250.
__device__ __forceinline__ unsigned short f32_to_bf16(float f) {
  union { __bf16 h; unsigned short u; } cv;
  cv.h = (__bf16)f;
  return cv.u;
}
__device__ __forceinline__ unsigned int pack_bf16x2(float x, float y) {
  union { __bf16 h[2]; unsigned int u; } cv;
  cv.h[0] = (__bf16)x;
  cv.h[1] = (__bf16)y;
  return cv.u;
}
__device__ __forceinline__ float bf16_to_f32(unsigned short h) {
  return __uint_as_float(((unsigned int)h) << 16);
}

union FragBF { v16bf v; uint4 q[2]; };

// ---- stage a 128x32 row-major tile (rows x k) into LDS as bf16 -------------
// bf16 source: async DMA straight into LDS (no VGPR round-trip) when available.
template<bool F32>
__device__ __forceinline__ void lds_load_rows(unsigned short (*dst)[LSTR],
    const void* __restrict__ src, long long row0, int col0, int ld) {
  const int t = threadIdx.x;
  if constexpr (F32) {
    const float* __restrict__ s = (const float*)src;
#pragma unroll
    for (int i = 0; i < 4; ++i) {
      int idx = t + i * 256;
      int r = idx >> 3;
      int c = (idx & 7) << 2;
      const float4 f = *(const float4*)(s + (row0 + r) * (long long)ld + col0 + c);
      *(uint2*)&dst[r][c] = make_uint2(pack_bf16x2(f.x, f.y), pack_bf16x2(f.z, f.w));
    }
  } else {
    const unsigned short* __restrict__ s = (const unsigned short*)src;
#pragma unroll
    for (int i = 0; i < 2; ++i) {
      int idx = t + i * 256;
      int r = idx >> 2;
      int c = (idx & 3) << 3;
      const unsigned short* g = s + (row0 + r) * (long long)ld + col0 + c;
#if HAVE_ASYNC_LDS
      __builtin_amdgcn_global_load_async_to_lds_b128(
          (gas_v4i_ptr)(void*)g, (las_v4i_ptr)(void*)&dst[r][c], 0, 0);
#else
      *(uint4*)&dst[r][c] = *(const uint4*)g;
#endif
    }
  }
}

// ---- stage a 32x128 (k x n) row-major tile into LDS transposed to (n,k) ----
template<bool F32>
__device__ __forceinline__ void lds_load_cols(unsigned short (*dst)[LSTR],
    const void* __restrict__ src, long long k0, int n0, int ld) {
  const int t = threadIdx.x;
  if constexpr (F32) {
    const float* __restrict__ s = (const float*)src;
#pragma unroll
    for (int i = 0; i < 4; ++i) {
      int idx = t + i * 256;
      int kr = idx >> 5;
      int c  = (idx & 31) << 2;
      const float4 f = *(const float4*)(s + (k0 + kr) * (long long)ld + n0 + c);
      unsigned int p01 = pack_bf16x2(f.x, f.y);
      unsigned int p23 = pack_bf16x2(f.z, f.w);
      dst[c + 0][kr] = (unsigned short)p01;
      dst[c + 1][kr] = (unsigned short)(p01 >> 16);
      dst[c + 2][kr] = (unsigned short)p23;
      dst[c + 3][kr] = (unsigned short)(p23 >> 16);
    }
  } else {
    const unsigned short* __restrict__ s = (const unsigned short*)src;
#pragma unroll
    for (int i = 0; i < 2; ++i) {
      int idx = t + i * 256;
      int kr = idx >> 4;
      int c  = (idx & 15) << 3;
      uint4 d = *(const uint4*)(s + (k0 + kr) * (long long)ld + n0 + c);
      unsigned int w[4] = {d.x, d.y, d.z, d.w};
#pragma unroll
      for (int j = 0; j < 4; ++j) {
        dst[c + 2 * j    ][kr] = (unsigned short)(w[j] & 0xFFFFu);
        dst[c + 2 * j + 1][kr] = (unsigned short)(w[j] >> 16);
      }
    }
  }
}

// ---- generic batched WMMA GEMM: C[bz] = A[bz] * B[bz] ----------------------
// A: M x K row-major (f32 or bf16). B_NT=true: B given as N x K row-major
// (e.g. K-matrix for Q*K^T). B_NT=false: B given as K x N row-major (W, V).
// C: bf16 or f32. Dimensions assumed multiples of the 128/128/32 tile.
template<bool A_F32, bool B_F32, bool B_NT, bool C_F32>
__global__ __launch_bounds__(256) void gemm_wmma(
    const void* __restrict__ Ap, const void* __restrict__ Bp, void* __restrict__ Cp,
    int K, int lda, int ldb, int ldc,
    long long sA, long long sB, long long sC) {
  __shared__ __align__(16) unsigned short As[2][BMT][LSTR];
  __shared__ __align__(16) unsigned short Bs[2][BNT][LSTR];

  // async staging is used whenever a bf16 tile goes through the row loader
  constexpr bool USE_ASYNC = (HAVE_ASYNC_LDS != 0) && ((!A_F32) || (B_NT && !B_F32));

  const int bz = blockIdx.z;
  const void* A = A_F32 ? (const void*)((const float*)Ap + bz * sA)
                        : (const void*)((const unsigned short*)Ap + bz * sA);
  const void* B = B_F32 ? (const void*)((const float*)Bp + bz * sB)
                        : (const void*)((const unsigned short*)Bp + bz * sB);

  const long long m0 = (long long)blockIdx.x * BMT;
  const long long n0 = (long long)blockIdx.y * BNT;

  const int lane = threadIdx.x & 31;
  const int wid  = threadIdx.x >> 5;
  const int wm   = wid >> 1;   // 4 waves along M (32 rows each)
  const int wn   = wid & 1;    // 2 waves along N (64 cols each)

  v8f acc[2][4];
#pragma unroll
  for (int i = 0; i < 2; ++i)
#pragma unroll
    for (int j = 0; j < 4; ++j)
#pragma unroll
      for (int e = 0; e < 8; ++e) acc[i][j][e] = 0.0f;

  const int mloc = lane & 15;
  const int kb   = (lane < 16) ? 0 : 8;   // 16-bit A/B fragment K-base per half-wave

  // Prologue: stage tile k0=0 into buffer 0.
  lds_load_rows<A_F32>(As[0], A, m0, 0, lda);
  if constexpr (B_NT) lds_load_rows<B_F32>(Bs[0], B, n0, 0, ldb);
  else                lds_load_cols<B_F32>(Bs[0], B, 0, (int)n0, ldb);
  if constexpr (USE_ASYNC) wait_async_lds();
  __syncthreads();

  for (int k0 = 0; k0 < K; k0 += BKT) {
    const int cur = (k0 >> 5) & 1;
    const int nxt = cur ^ 1;

    // Stage next tile while this tile computes (double buffering).
    if (k0 + BKT < K) {
      lds_load_rows<A_F32>(As[nxt], A, m0, k0 + BKT, lda);
      if constexpr (B_NT) lds_load_rows<B_F32>(Bs[nxt], B, n0, k0 + BKT, ldb);
      else                lds_load_cols<B_F32>(Bs[nxt], B, k0 + BKT, (int)n0, ldb);
      // Speculative prefetch one tile further ahead (global_prefetch_b8).
      if (k0 + 2 * BKT < K) {
        long long pr = m0 + (threadIdx.x >> 1);
        int pc = k0 + 2 * BKT + ((threadIdx.x & 1) << 4);
        if constexpr (A_F32)
          __builtin_prefetch((const float*)A + pr * (long long)lda + pc, 0, 1);
        else
          __builtin_prefetch((const unsigned short*)A + pr * (long long)lda + pc, 0, 1);
      }
    }

    // Compute on the current buffer.
    FragBF a[2], b[4];
#pragma unroll
    for (int mi = 0; mi < 2; ++mi) {
      int r = wm * 32 + mi * 16 + mloc;
      a[mi].q[0] = *(const uint4*)&As[cur][r][kb];        // K = kb .. kb+7
      a[mi].q[1] = *(const uint4*)&As[cur][r][kb + 16];   // K = kb+16 .. kb+23
    }
#pragma unroll
    for (int ni = 0; ni < 4; ++ni) {
      int r = wn * 64 + ni * 16 + mloc;
      b[ni].q[0] = *(const uint4*)&Bs[cur][r][kb];
      b[ni].q[1] = *(const uint4*)&Bs[cur][r][kb + 16];
    }
#pragma unroll
    for (int mi = 0; mi < 2; ++mi)
#pragma unroll
      for (int ni = 0; ni < 4; ++ni)
        acc[mi][ni] = __builtin_amdgcn_wmma_f32_16x16x32_bf16(
            false, a[mi].v, false, b[ni].v, (short)0, acc[mi][ni], false, false);

    if constexpr (USE_ASYNC) wait_async_lds();
    __syncthreads();
  }

  // Epilogue: C/D 16x16 f32 layout: lanes 0-15 -> N=lane, M=vgpr;
  // lanes 16-31 -> N=lane-16, M=vgpr+8.
  const int cn = lane & 15;
  const int mo = (lane >> 4) * 8;
#pragma unroll
  for (int mi = 0; mi < 2; ++mi)
#pragma unroll
    for (int ni = 0; ni < 4; ++ni) {
      long long rowb = m0 + wm * 32 + mi * 16 + mo;
      long long colb = n0 + wn * 64 + ni * 16 + cn;
      if constexpr (C_F32) {
        float* C = (float*)Cp + bz * sC;
#pragma unroll
        for (int r = 0; r < 8; ++r)
          C[(rowb + r) * (long long)ldc + colb] = acc[mi][ni][r];
      } else {
        unsigned short* C = (unsigned short*)Cp + bz * sC;
#pragma unroll
        for (int r = 0; r < 8; ++r)
          C[(rowb + r) * (long long)ldc + colb] = f32_to_bf16(acc[mi][ni][r]);
      }
    }
}

// ---- XPOS rotary (torchscale variant), in place on bf16, one thread/pair ---
__global__ __launch_bounds__(256) void xpos_kernel(unsigned short* __restrict__ buf,
                                                   int L, int D, int downscale) {
  const int half = D >> 1;
  long long i = (long long)blockIdx.x * blockDim.x + threadIdx.x; // pair index
  int j = (int)(i % half);
  int l = (int)((i / half) % L);
  unsigned short* p = buf + 2 * i;
  float xe = bf16_to_f32(p[0]);
  float xo = bf16_to_f32(p[1]);
  float sv    = (2.0f * j + 0.4f * D) / (1.4f * D);
  float power = (float)l * (1.0f / 512.0f);          // SCALE_BASE
  float scale = __expf(power * __logf(sv));
  if (downscale) scale = 1.0f / scale;
  float ang = (float)l * __expf((float)j * (-9.210340371976184f / half)); // 10000^{-j/half}
  float sn = __sinf(ang) * scale;
  float cs = __cosf(ang) * scale;
  p[0] = f32_to_bf16(xe * cs - xo * sn);
  p[1] = f32_to_bf16(xo * cs + xe * sn);
}

// ---- softmax over k + decay gamma^|q-k|, in place on bf16 scores -----------
__global__ __launch_bounds__(256) void softmax_decay_kernel(unsigned short* __restrict__ S,
                                                            int L) {
  __shared__ float red[8];
  const long long row = blockIdx.x;
  const int q = (int)(row % L);
  unsigned short* s = S + row * (long long)L;
  const int lane = threadIdx.x & 31, wid = threadIdx.x >> 5;

  float v[8];
  float m = -3.4e38f;
#pragma unroll
  for (int i = 0; i < 8; ++i) {
    v[i] = bf16_to_f32(s[threadIdx.x + i * 256]);
    m = fmaxf(m, v[i]);
  }
#pragma unroll
  for (int o = 16; o > 0; o >>= 1) m = fmaxf(m, __shfl_xor(m, o, 32));
  if (lane == 0) red[wid] = m;
  __syncthreads();
  m = fmaxf(fmaxf(fmaxf(red[0], red[1]), fmaxf(red[2], red[3])),
            fmaxf(fmaxf(red[4], red[5]), fmaxf(red[6], red[7])));

  float Z = 0.0f;
#pragma unroll
  for (int i = 0; i < 8; ++i) { v[i] = __expf(v[i] - m); Z += v[i]; }
#pragma unroll
  for (int o = 16; o > 0; o >>= 1) Z += __shfl_xor(Z, o, 32);
  __syncthreads();
  if (lane == 0) red[wid] = Z;
  __syncthreads();
  Z = red[0] + red[1] + red[2] + red[3] + red[4] + red[5] + red[6] + red[7];

  const float invZ = 1.0f / Z;
  const float lng  = -0.10536051565782628f;   // ln(0.9)
#pragma unroll
  for (int i = 0; i < 8; ++i) {
    int k = threadIdx.x + i * 256;
    float p = v[i] * invZ * __expf(lng * fabsf((float)(q - k)));
    s[k] = f32_to_bf16(p);
  }
}

extern "C" void kernel_launch(void* const* d_in, const int* in_sizes, int n_in,
                              void* d_out, int out_size, void* d_ws, size_t ws_size,
                              hipStream_t stream) {
  (void)in_sizes; (void)n_in; (void)out_size; (void)ws_size;
  const int B = 8, L = 2048, H = 1024, Dh = 1024;
  const float* X  = (const float*)d_in[0];
  const float* WQ = (const float*)d_in[1];
  const float* WK = (const float*)d_in[2];
  const float* WV = (const float*)d_in[3];
  float* out = (float*)d_out;

  const long long nQKV = (long long)B * L * Dh;   // per-buffer elements
  unsigned short* Qb = (unsigned short*)d_ws;
  unsigned short* Kb = Qb + nQKV;
  unsigned short* Vb = Kb + nQKV;
  unsigned short* Sb = Vb + nQKV;                 // B*L*L bf16 scores / probs

  dim3 blk(256, 1, 1);

  // 1) Projections: C(16384x1024) = X(16384x1024) * W(1024x1024), bf16 out.
  {
    dim3 g((B * L) / BMT, Dh / BNT, 1);
    gemm_wmma<true, true, false, false><<<g, blk, 0, stream>>>(X, WQ, Qb, H, H, Dh, Dh, 0, 0, 0);
    gemm_wmma<true, true, false, false><<<g, blk, 0, stream>>>(X, WK, Kb, H, H, Dh, Dh, 0, 0, 0);
    gemm_wmma<true, true, false, false><<<g, blk, 0, stream>>>(X, WV, Vb, H, H, Dh, Dh, 0, 0, 0);
  }

  // 2) XPOS rotary on Q (and inverse scale on K).
  {
    long long pairs = (long long)B * L * (Dh / 2);
    dim3 g((unsigned)(pairs / 256), 1, 1);
    xpos_kernel<<<g, blk, 0, stream>>>(Qb, L, Dh, 0);
    xpos_kernel<<<g, blk, 0, stream>>>(Kb, L, Dh, 1);
  }

  // 3) Scores: S[b] = Q[b] (2048x1024) * K[b]^T, bf16 out. B is N x K rows (NT).
  {
    dim3 g(L / BMT, L / BNT, B);
    gemm_wmma<false, false, true, false><<<g, blk, 0, stream>>>(
        Qb, Kb, Sb, Dh, Dh, Dh, L,
        (long long)L * Dh, (long long)L * Dh, (long long)L * L);
  }

  // 4) softmax over k, times gamma^|q-k|, in place.
  {
    dim3 g((unsigned)(B * L), 1, 1);
    softmax_decay_kernel<<<g, blk, 0, stream>>>(Sb, L);
  }

  // 5) out[b] = P[b] (2048x2048) * V[b] (2048x1024), f32 out straight to d_out.
  {
    dim3 g(L / BMT, Dh / BNT, B);
    gemm_wmma<false, false, false, true><<<g, blk, 0, stream>>>(
        Sb, Vb, out, L, L, Dh, Dh,
        (long long)L * L, (long long)L * Dh, (long long)L * Dh);
  }
}
